// RoPETransformerEncoderLayer_19310172963547
// MI455X (gfx1250) — matrix operand order repaired and checked by reading, not compiled, not run on hardware
//
#include <hip/hip_runtime.h>

#define DEV __device__ __forceinline__

typedef __bf16 bf16t;
typedef bf16t v16bf __attribute__((ext_vector_type(16)));
typedef float v8f  __attribute__((ext_vector_type(8)));

union FragA { v16bf v; unsigned short u[16]; };
union FragB { v16bf v; unsigned short u[16]; };

DEV unsigned short f2bf(float f) {
    union { float f; unsigned u; } c; c.f = f;
    unsigned r = c.u + 0x7FFFu + ((c.u >> 16) & 1u);   // RNE
    return (unsigned short)(r >> 16);
}

DEV v8f wmma_bf16(v16bf a, v16bf b, v8f c) {
    return __builtin_amdgcn_wmma_f32_16x16x32_bf16(
        false, a, false, b, (short)0, c, false, false);
}

// A-matrix 16x32 bf16 element k-index for element e, lane-half hi (ISA 7.12.2)
DEV int akmap(int e, int hi) {
    int v = e >> 1, p = e & 1;
    return ((v & 4) ? 16 : 0) + hi * 8 + ((v & 3) << 1) + p;
}

DEV v8f vzero8() { v8f z; for (int i = 0; i < 8; ++i) z[i] = 0.0f; return z; }

// xor-lane swizzle within groups of 32 (and_mask=0x1f, or=0, xor=m)
#define FSWZ(v, m) __uint_as_float(__builtin_amdgcn_ds_swizzle(__float_as_uint(v), ((m) << 10) | 0x1f))

// async global->LDS 16B per lane (GVS form), portable across toolchains via asm
#define ASYNC_B128(ldsOff, gByteOff, sBase)                                   \
    asm volatile("global_load_async_to_lds_b128 %0, %1, %2"                   \
                 :: "v"(ldsOff), "v"(gByteOff), "s"(sBase) : "memory")
#define WAIT_ASYNC() asm volatile("s_wait_asynccnt 0" ::: "memory")

// ---------------------------------------------------------------- cast f32->bf16 (plain)
__global__ void k_cast_bf16(const float* __restrict__ in, unsigned short* __restrict__ out, int n) {
    int i = blockIdx.x * 256 + threadIdx.x;
    if (i < n) out[i] = f2bf(in[i]);
}

// ---------------------------------------------------------------- cast f32->bf16, transpose [K,N]->[N,K]
__global__ void k_castT_bf16(const float* __restrict__ in, unsigned short* __restrict__ out,
                             int K, int N) {
    int idx = blockIdx.x * 256 + threadIdx.x;      // K*N threads
    int n = idx / K, k = idx % K;
    out[idx] = f2bf(in[(size_t)k * N + n]);        // out[n*K + k]
}

// ---------------------------------------------------------------- layernorm row=1024 -> bf16
__global__ __launch_bounds__(256) void k_ln_bf16(const float* __restrict__ x,
                                                 const float* __restrict__ g,
                                                 const float* __restrict__ b,
                                                 unsigned short* __restrict__ out) {
    __shared__ float part[8];
    const int row = blockIdx.x, tid = threadIdx.x;
    const int wave = tid >> 5, lane = tid & 31;
    const float* xr = x + (size_t)row * 1024;
    float v[4], s = 0.0f;
#pragma unroll
    for (int i = 0; i < 4; ++i) { v[i] = xr[tid + i * 256]; s += v[i]; }
    s += FSWZ(s, 16); s += FSWZ(s, 8); s += FSWZ(s, 4); s += FSWZ(s, 2); s += FSWZ(s, 1);
    if (!lane) part[wave] = s;
    __syncthreads();
    float tot = 0.0f;
#pragma unroll
    for (int j = 0; j < 8; ++j) tot += part[j];
    const float mu = tot * (1.0f / 1024.0f);
    float ss = 0.0f;
#pragma unroll
    for (int i = 0; i < 4; ++i) { float d = v[i] - mu; ss += d * d; }
    ss += FSWZ(ss, 16); ss += FSWZ(ss, 8); ss += FSWZ(ss, 4); ss += FSWZ(ss, 2); ss += FSWZ(ss, 1);
    __syncthreads();
    if (!lane) part[wave] = ss;
    __syncthreads();
    float vt = 0.0f;
#pragma unroll
    for (int j = 0; j < 8; ++j) vt += part[j];
    const float rstd = rsqrtf(vt * (1.0f / 1024.0f) + 1e-5f);
#pragma unroll
    for (int i = 0; i < 4; ++i) {
        int c = tid + i * 256;
        out[(size_t)row * 1024 + c] = f2bf((v[i] - mu) * rstd * g[c] + b[c]);
    }
}

// ---------------------------------------------------------------- WMMA GEMM 128x128 tile
// A: [M,K] bf16 row-major; Bt: [N,K] bf16 (pre-transposed weights).
// EPI: 0 = +bias -> f32 ; 1 = +bias +residual -> f32 ; 2 = +bias, exact GELU -> bf16
template <int EPI>
__global__ __launch_bounds__(256) void k_gemm_bf16(const unsigned short* __restrict__ A,
                                                   const unsigned short* __restrict__ Bt,
                                                   const float* __restrict__ bias,
                                                   const float* __restrict__ resid,
                                                   float* __restrict__ outF,
                                                   unsigned short* __restrict__ outB,
                                                   int M, int N, int K) {
    __shared__ __align__(16) unsigned short sA [128 * 40];  // [row][k]  pad 40
    __shared__ __align__(16) unsigned short sBt[128 * 40];  // [col][k]  pad 40
    const int tid = threadIdx.x;
    const int wave = tid >> 5, lane = tid & 31;
    const int wm = wave & 3, wn = wave >> 2;                // 4x2 wave grid, 32x64 per wave
    const int r16 = lane & 15, hi = lane >> 4;
    const size_t row0 = (size_t)blockIdx.y * 128;
    const size_t col0 = (size_t)blockIdx.x * 128;
    const unsigned sAb = (unsigned)(unsigned long long)&sA[0];
    const unsigned sBb = (unsigned)(unsigned long long)&sBt[0];

    v8f acc[2][4];
#pragma unroll
    for (int mi = 0; mi < 2; ++mi)
#pragma unroll
        for (int ni = 0; ni < 4; ++ni) acc[mi][ni] = vzero8();

    for (int k0 = 0; k0 < K; k0 += 32) {
        // both 128x32 bf16 tiles stream straight into LDS via async DMA (16B/lane)
#pragma unroll
        for (int i = tid; i < 512; i += 256) {
            int r = i >> 2, q = i & 3;
            unsigned lo = r * 80 + q * 16;
            ASYNC_B128(sAb + lo, (unsigned)((((row0 + r) * (size_t)K) + k0 + q * 8) * 2), A);
            ASYNC_B128(sBb + lo, (unsigned)((((col0 + r) * (size_t)K) + k0 + q * 8) * 2), Bt);
        }
        if (k0 + 32 < K) {                                   // global_prefetch_b8
            __builtin_prefetch(A  + (row0 + (tid >> 1)) * (size_t)K + k0 + 32, 0, 1);
            __builtin_prefetch(Bt + (col0 + (tid >> 1)) * (size_t)K + k0 + 32, 0, 1);
        }
        WAIT_ASYNC();
        __syncthreads();

        FragA af[2]; FragB bfr[4];
#pragma unroll
        for (int mi = 0; mi < 2; ++mi) {
            int rb = (wm * 32 + mi * 16 + r16) * 40;
#pragma unroll
            for (int e = 0; e < 16; ++e) af[mi].u[e] = sA[rb + hi * 16 + e];
        }
#pragma unroll
        for (int ni = 0; ni < 4; ++ni) {
            int cb = (wn * 64 + ni * 16 + r16) * 40 + hi * 16;   // contiguous 16 shorts
#pragma unroll
            for (int e = 0; e < 16; ++e) bfr[ni].u[e] = sBt[cb + e];
        }
#pragma unroll
        for (int mi = 0; mi < 2; ++mi)
#pragma unroll
            for (int ni = 0; ni < 4; ++ni)
                acc[mi][ni] = wmma_bf16(af[mi].v, bfr[ni].v, acc[mi][ni]);
        __syncthreads();
    }

#pragma unroll
    for (int mi = 0; mi < 2; ++mi)
#pragma unroll
        for (int ni = 0; ni < 4; ++ni)
#pragma unroll
            for (int i = 0; i < 8; ++i) {
                size_t gr = row0 + wm * 32 + mi * 16 + i + 8 * hi;
                size_t gc = col0 + wn * 64 + ni * 16 + r16;
                float v = acc[mi][ni][i] + bias[gc];
                size_t idx = gr * (size_t)N + gc;
                if (EPI == 1)      outF[idx] = v + resid[idx];
                else if (EPI == 2) outB[idx] = f2bf(0.5f * v * (1.0f + erff(v * 0.70710678f)));
                else               outF[idx] = v;
            }
}

// ---------------------------------------------------------------- RoPE + head split
// Q,K -> bf16 [B,H,T,HD] ; V -> bf16 transposed [B,H,HD,T]
__global__ void k_rope_split(const float* __restrict__ qkv,
                             const float* __restrict__ cs, const float* __restrict__ sn,
                             unsigned short* __restrict__ Qo, unsigned short* __restrict__ Ko,
                             unsigned short* __restrict__ Vt) {
    unsigned idx = blockIdx.x * 256 + threadIdx.x;   // B*T*H*32 = 2097152 threads
    int d = idx & 31;
    int h = (idx >> 5) & 15;
    int row = idx >> 9;                               // b*T + t, 0..4095
    int t = row & 2047;
    int b = row >> 11;
    const float* p = qkv + (size_t)row * 3072 + h * 64;
    float c = cs[t * 32 + d], s = sn[t * 32 + d];
    float q1 = p[2 * d], q2 = p[2 * d + 1];
    float k1 = p[1024 + 2 * d], k2 = p[1024 + 2 * d + 1];
    float v1 = p[2048 + 2 * d], v2 = p[2048 + 2 * d + 1];
    size_t bh = (size_t)(b * 16 + h);
    size_t ob = (bh * 2048 + t) * 64;
    Qo[ob + d]      = f2bf(q1 * c - q2 * s);
    Qo[ob + 32 + d] = f2bf(q1 * s + q2 * c);
    Ko[ob + d]      = f2bf(k1 * c - k2 * s);
    Ko[ob + 32 + d] = f2bf(k1 * s + k2 * c);
    size_t vb = bh * 64 * 2048 + t;                   // Vt[bh][dim][t]
    Vt[vb + (size_t)(2 * d)     * 2048] = f2bf(v1);
    Vt[vb + (size_t)(2 * d + 1) * 2048] = f2bf(v2);
}

// ---------------------------------------------------------------- flash attention (WMMA, async-to-LDS K/V)
__global__ __launch_bounds__(256) void k_attn(const unsigned short* __restrict__ Q,
                                              const unsigned short* __restrict__ Kt,
                                              const unsigned short* __restrict__ Vt,
                                              unsigned short* __restrict__ Out) {
    __shared__ __align__(16) unsigned short sK [32 * 72];      // [key][dim]   (+pad)
    __shared__ __align__(16) unsigned short sVt[64 * 40];      // [dim][key]   (+pad)
    __shared__ __align__(16) unsigned short sP [8 * 16 * 36];  // per-wave 16x32 P tile (+pad)
    const int tid = threadIdx.x, wave = tid >> 5, lane = tid & 31;
    const int r16 = lane & 15, hi = lane >> 4;
    const int bh = blockIdx.y;
    const int qbase = blockIdx.x * 128 + wave * 16;
    const size_t hb = (size_t)bh * 2048 * 64;

    FragA qa[2];                                               // Q rows held resident (16x64)
#pragma unroll
    for (int c = 0; c < 2; ++c)
#pragma unroll
        for (int e = 0; e < 16; ++e)
            qa[c].u[e] = Q[hb + (size_t)(qbase + r16) * 64 + c * 32 + hi * 16 + e];

    v8f o[4];
#pragma unroll
    for (int d = 0; d < 4; ++d) o[d] = vzero8();
    float mrow[8], lrow[8];
#pragma unroll
    for (int i = 0; i < 8; ++i) { mrow[i] = -1e30f; lrow[i] = 0.0f; }

    const unsigned sKb = (unsigned)(unsigned long long)&sK[0];
    const unsigned sVb = (unsigned)(unsigned long long)&sVt[0];
    const int kr = tid >> 3, kq = tid & 7;                     // K tile: 32 keys x 64 dims
    const int vr = tid >> 2, vq = tid & 3;                     // V tile: 64 dims x 32 keys
    const unsigned short* kp = Kt + hb;
    const unsigned short* vp = Vt + hb;

    for (int kc = 0; kc < 2048; kc += 32) {
        ASYNC_B128(sKb + kr * 144 + kq * 16,
                   (unsigned)((((kc + kr) * 64) + kq * 8) * 2), kp);
        ASYNC_B128(sVb + vr * 80 + vq * 16,
                   (unsigned)(((vr * 2048) + kc + vq * 8) * 2), vp);
        WAIT_ASYNC();
        __syncthreads();

        // scores S = (Q K^T) for 16 queries x 32 keys: 2 N-tiles, K-dim 64 = 2 WMMAs each
        v8f s2[2];
#pragma unroll
        for (int j = 0; j < 2; ++j) {
            FragB bk0, bk1;
#pragma unroll
            for (int e = 0; e < 16; ++e) {
                bk0.u[e] = sK[(16 * j + r16) * 72 +      hi * 16 + e];
                bk1.u[e] = sK[(16 * j + r16) * 72 + 32 + hi * 16 + e];
            }
            v8f z = vzero8();
            s2[j] = wmma_bf16(qa[0].v, bk0.v, z);
            s2[j] = wmma_bf16(qa[1].v, bk1.v, s2[j]);
        }

        // online softmax; row m = i + 8*hi lives across the 16 lanes of each half-wave
        const float SC = 0.125f;                              // 1/sqrt(64)
        const unsigned pb = wave * 576;
#pragma unroll
        for (int i = 0; i < 8; ++i) {
            float v0 = s2[0][i] * SC, v1 = s2[1][i] * SC;
            float mx = fmaxf(v0, v1);
            mx = fmaxf(mx, FSWZ(mx, 1)); mx = fmaxf(mx, FSWZ(mx, 2));
            mx = fmaxf(mx, FSWZ(mx, 4)); mx = fmaxf(mx, FSWZ(mx, 8));
            float mn = fmaxf(mrow[i], mx);
            float p0 = __expf(v0 - mn), p1 = __expf(v1 - mn);
            float rs = p0 + p1;
            rs += FSWZ(rs, 1); rs += FSWZ(rs, 2); rs += FSWZ(rs, 4); rs += FSWZ(rs, 8);
            float corr = __expf(mrow[i] - mn);
            lrow[i] = lrow[i] * corr + rs;
            mrow[i] = mn;
#pragma unroll
            for (int d = 0; d < 4; ++d) o[d][i] *= corr;
            int m = i + 8 * hi;
            sP[pb + m * 36 + r16]      = f2bf(p0);
            sP[pb + m * 36 + 16 + r16] = f2bf(p1);
        }

        // O += P(16x32) @ V(32x64); V fragments are contiguous b128 runs in sVt
        FragA pa;
#pragma unroll
        for (int e = 0; e < 16; ++e) pa.u[e] = sP[pb + r16 * 36 + hi * 16 + e];
#pragma unroll
        for (int d = 0; d < 4; ++d) {
            FragB bv;
            int vb = (d * 16 + r16) * 40 + hi * 16;            // contiguous 16 shorts
#pragma unroll
            for (int e = 0; e < 16; ++e) bv.u[e] = sVt[vb + e];
            o[d] = wmma_bf16(pa.v, bv.v, o[d]);
        }
        __syncthreads();
    }

    const int b = bh >> 4, h = bh & 15;
#pragma unroll
    for (int i = 0; i < 8; ++i) {
        float inv = 1.0f / lrow[i];
        int t = qbase + i + 8 * hi;
        size_t ro = ((size_t)b * 2048 + t) * 1024 + h * 64;
#pragma unroll
        for (int d = 0; d < 4; ++d)
            Out[ro + d * 16 + r16] = f2bf(o[d][i] * inv);
    }
}

// ---------------------------------------------------------------- launch
extern "C" void kernel_launch(void* const* d_in, const int* in_sizes, int n_in,
                              void* d_out, int out_size, void* d_ws, size_t ws_size,
                              hipStream_t stream) {
    const float* x    = (const float*)d_in[0];
    const float* ln1g = (const float*)d_in[1];
    const float* ln1b = (const float*)d_in[2];
    const float* ln2g = (const float*)d_in[3];
    const float* ln2b = (const float*)d_in[4];
    const float* Wqkv = (const float*)d_in[5];
    const float* bqkv = (const float*)d_in[6];
    const float* Wo   = (const float*)d_in[7];
    const float* bo   = (const float*)d_in[8];
    const float* W1   = (const float*)d_in[9];
    const float* b1   = (const float*)d_in[10];
    const float* W2   = (const float*)d_in[11];
    const float* b2   = (const float*)d_in[12];
    const float* cosT = (const float*)d_in[13];
    const float* sinT = (const float*)d_in[14];
    float* out = (float*)d_out;

    char* ws = (char*)d_ws;
    size_t off = 0;
    auto alloc = [&](size_t bytes) -> void* {
        void* p = ws + off; off += (bytes + 255) & ~(size_t)255; return p;
    };
    const size_t BT = 4096;  // B*T
    unsigned short* wqkv_b = (unsigned short*)alloc((size_t)1024 * 3072 * 2);  // [N,K]
    unsigned short* wo_b   = (unsigned short*)alloc((size_t)1024 * 1024 * 2);  // [N,K]
    unsigned short* w1_b   = (unsigned short*)alloc((size_t)1024 * 4096 * 2);  // [N,K]
    unsigned short* w2_b   = (unsigned short*)alloc((size_t)4096 * 1024 * 2);  // [N,K]
    unsigned short* xn_b   = (unsigned short*)alloc(BT * 1024 * 2);
    float*          qkv    = (float*)         alloc(BT * 3072 * 4);
    unsigned short* q_b    = (unsigned short*)alloc(BT * 1024 * 2);
    unsigned short* k_b    = (unsigned short*)alloc(BT * 1024 * 2);
    unsigned short* vt_b   = (unsigned short*)alloc(BT * 1024 * 2);
    unsigned short* ao_b   = (unsigned short*)alloc(BT * 1024 * 2);
    float*          x2     = (float*)         alloc(BT * 1024 * 4);
    unsigned short* h_b    = (unsigned short*)alloc(BT * 1024 * 2);
    unsigned short* g_b    = (unsigned short*)alloc(BT * 4096 * 2);

    // weights -> bf16, transposed to [N,K] for async-DMA friendly GEMM B tiles
    k_castT_bf16<<<(1024 * 3072) / 256, 256, 0, stream>>>(Wqkv, wqkv_b, 1024, 3072);
    k_castT_bf16<<<(1024 * 1024) / 256, 256, 0, stream>>>(Wo,   wo_b,   1024, 1024);
    k_castT_bf16<<<(1024 * 4096) / 256, 256, 0, stream>>>(W1,   w1_b,   1024, 4096);
    k_castT_bf16<<<(4096 * 1024) / 256, 256, 0, stream>>>(W2,   w2_b,   4096, 1024);

    // LN1
    k_ln_bf16<<<4096, 256, 0, stream>>>(x, ln1g, ln1b, xn_b);
    // QKV projection
    k_gemm_bf16<0><<<dim3(3072 / 128, 4096 / 128), 256, 0, stream>>>(
        xn_b, wqkv_b, bqkv, nullptr, qkv, nullptr, 4096, 3072, 1024);
    // RoPE + split heads (V transposed)
    k_rope_split<<<(4096 * 16 * 32) / 256, 256, 0, stream>>>(qkv, cosT, sinT, q_b, k_b, vt_b);
    // attention
    k_attn<<<dim3(2048 / 128, 32), 256, 0, stream>>>(q_b, k_b, vt_b, ao_b);
    // output projection + residual
    k_gemm_bf16<1><<<dim3(1024 / 128, 4096 / 128), 256, 0, stream>>>(
        ao_b, wo_b, bo, x, x2, nullptr, 4096, 1024, 1024);
    // LN2
    k_ln_bf16<<<4096, 256, 0, stream>>>(x2, ln2g, ln2b, h_b);
    // FFN1 + exact GELU
    k_gemm_bf16<2><<<dim3(4096 / 128, 4096 / 128), 256, 0, stream>>>(
        h_b, w1_b, b1, nullptr, nullptr, g_b, 4096, 4096, 1024);
    // FFN2 + residual -> output
    k_gemm_bf16<1><<<dim3(1024 / 128, 4096 / 128), 256, 0, stream>>>(
        g_b, w2_b, b2, x2, out, nullptr, 4096, 1024, 4096);
}
